// Convolution_58171037057365
// MI455X (gfx1250) — compile-verified, matrix operand order checked by reading, not run on hardware
//
#include <hip/hip_runtime.h>
#include <hip/hip_bf16.h>

// ---------------------------------------------------------------------------
// MI455X (gfx1250) implicit-GEMM equivariant conv.
//   out = sc + conv3d(x, K(tp_weight))  collapses to a single conv whose
//   kernel has the self-connection folded into the center tap.
// Compute: 227 GFLOP, bytes ~113MB -> AI ~2000 FLOP/B -> pure compute bound.
// Precision: bf16x3 split (hi*hi + hi*lo + lo*hi) with f32 accumulation via
// v_wmma_f32_16x16x32_bf16 (codegen-confirmed builtin on this toolchain).
// ---------------------------------------------------------------------------

typedef __attribute__((ext_vector_type(8)))  float  v8f;
typedef __attribute__((ext_vector_type(16))) __bf16 v16bf;

union U32x8 {
    unsigned int u[8];
    uint4        q[2];
    v16bf        v;
};

__device__ __forceinline__ unsigned short f32_to_bf16_rne(float f) {
    unsigned int u = __float_as_uint(f);
    unsigned int r = u + 0x7FFFu + ((u >> 16) & 1u);
    return (unsigned short)(r >> 16);
}
__device__ __forceinline__ float bf16_bits_to_f32(unsigned short h) {
    return __uint_as_float(((unsigned int)h) << 16);
}

// ---------------------------------------------------------------------------
// Constants from the reference
// ---------------------------------------------------------------------------
#define MUL        16
#define KS         5
#define NTAP       125
#define GRID       48
#define CH         64
#define CH_STRIDE  (48 * 48 * 48)        // 110592 floats per channel
#define PW0        0.17677669529663687f  // sqrt(1/32)
#define PW1        0.30618621784789724f  // sqrt(3/32)
#define INV_SQRT3  0.57735026918962576f
#define Y0_CONST   0.28209479177387814f  // 1/(2*sqrt(pi))
#define Y1_CONST   0.48860251190291992f  // sqrt(3/(4*pi))

// B-stream layout in workspace: per tap t (125), chunk c (2), n-subtile nt (4):
// one WMMA B fragment = 32 lanes x 8 dwords, stored lane-contiguous so the
// GEMM reads it as two aligned b128 loads per lane.
//   Bhi[((t*2+c)*4+nt)*256 + lane*8 + j]
#define FRAG_DW    256
#define NFRAG      (NTAP * 8)            // 1000 fragments
#define B_DWORDS   (NFRAG * FRAG_DW)     // 256000 dwords = 1 MB per stream

// ---------------------------------------------------------------------------
// Stage 1: build the folded conv kernel, pre-swizzled into WMMA-B fragments.
// One block per tap, 256 threads.
// ---------------------------------------------------------------------------
__global__ __launch_bounds__(256) void build_kernel_frags(
    const float* __restrict__ w_sc0,   // [16,16]
    const float* __restrict__ w_sc1,   // [16,16]
    const float* __restrict__ tp,      // [5,1024]
    unsigned int* __restrict__ Bhi,
    unsigned int* __restrict__ Blo)
{
    __shared__ float Mt[CH * CH];      // M[cin][cout] for this tap (16 KB)

    const int t   = blockIdx.x;        // tap = (kd*5 + kh)*5 + kw
    const int tid = threadIdx.x;
    const int kd = t / 25, kh = (t / 5) % 5, kw = t % 5;

    // Lattice geometry for this tap
    const float lx = (float)(kd - 2), ly = (float)(kh - 2), lz = (float)(kw - 2);
    const float dn = sqrtf(lx * lx + ly * ly + lz * lz);
    const float safe = (dn > 0.f) ? dn : 1.f;
    float ym[3];
    ym[0] = Y1_CONST * lx / safe;
    ym[1] = Y1_CONST * ly / safe;
    ym[2] = Y1_CONST * lz / safe;

    // Radial embedding emb[r] = C * sus(diff+1) * sus(1-diff),
    // diff = d/step - (r+1), step = (D/2)/6 -> 1/step = 2.4
    float embv[5];
    const float C0 = 1.14136f * expf(2.0f);
    #pragma unroll
    for (int r = 0; r < 5; ++r) {
        float diff = 2.4f * dn - (float)(r + 1);
        float a = diff + 1.0f;
        float b = 1.0f - diff;
        float sa = (a > 0.f) ? expf(-1.0f / a) : 0.f;
        float sb = (b > 0.f) ? expf(-1.0f / b) : 0.f;
        embv[r] = C0 * sa * sb;
    }

    // W(i,u,w) = (emb @ tp)/125 at this tap
    auto Wf = [&](int i, int u, int w) -> float {
        int o = ((i * 16 + u) << 4) + w;
        return 0.008f * (embv[0] * tp[o] +
                         embv[1] * tp[1024 + o] +
                         embv[2] * tp[2048 + o] +
                         embv[3] * tp[3072 + o] +
                         embv[4] * tp[4096 + o]);
    };

    // Each thread fills 16 entries of the 64x64 tile (cin-major).
    #pragma unroll
    for (int e = 0; e < 16; ++e) {
        int lin  = (tid << 4) + e;
        int cin  = lin >> 6;
        int cout = lin & 63;
        float v;
        if (cin < 16) {
            if (cout < 16) {
                v = PW0 * Y0_CONST * Wf(0, cin, cout);                 // Kss (W00)
            } else {
                int co = cout - 16;
                v = PW1 * INV_SQRT3 * Wf(1, cin, co / 3) * ym[co % 3]; // Ksv (W01)
            }
        } else {
            int u = (cin - 16) / 3, m = (cin - 16) % 3;
            if (cout < 16) {
                v = PW0 * INV_SQRT3 * Wf(3, u, cout) * ym[m];          // Kvs (W11)
            } else {
                int w = (cout - 16) / 3, n = (cout - 16) % 3;
                v = (m == n) ? (PW1 * INV_SQRT3 * Wf(2, u, w) * Y0_CONST) : 0.f; // Kvv (W10)
            }
        }
        // Fold self-connection into the center tap (kd=kh=kw=2 -> t==62)
        if (t == 62) {
            if (cin < 16 && cout < 16) {
                v += 0.25f * w_sc0[cin * 16 + cout];
            } else if (cin >= 16 && cout >= 16) {
                int u = (cin - 16) / 3, m = (cin - 16) % 3;
                int w = (cout - 16) / 3, n = (cout - 16) % 3;
                if (m == n) v += 0.25f * w_sc1[u * 16 + w];
            }
        }
        Mt[lin] = v;
    }
    __syncthreads();

    // Swizzle into WMMA B fragments (ISA 7.12.2: lane = N column; VGPR j holds
    // K pair k0 = c*32 + (j>=4?16:0) + (lane>=16?8:0) + (j&3)*2; low 16 = k0).
    const int fi    = tid >> 5;            // 0..7 : chunk c = fi>>2, nt = fi&3
    const int lane  = tid & 31;
    const int c     = fi >> 2;
    const int nt    = fi & 3;
    const int n     = nt * 16 + (lane & 15);
    const int khalf = (lane >> 4) << 3;
    const unsigned base = ((unsigned)t * 8 + fi) * FRAG_DW + (lane << 3);
    #pragma unroll
    for (int j = 0; j < 8; ++j) {
        int k0 = c * 32 + ((j & 4) << 2) + khalf + ((j & 3) << 1);
        float v0 = Mt[k0 * 64 + n];
        float v1 = Mt[(k0 + 1) * 64 + n];
        unsigned short h0 = f32_to_bf16_rne(v0);
        unsigned short h1 = f32_to_bf16_rne(v1);
        Bhi[base + j] = ((unsigned)h1 << 16) | (unsigned)h0;
        unsigned short l0 = f32_to_bf16_rne(v0 - bf16_bits_to_f32(h0));
        unsigned short l1 = f32_to_bf16_rne(v1 - bf16_bits_to_f32(h1));
        Blo[base + j] = ((unsigned)l1 << 16) | (unsigned)l0;
    }
}

// ---------------------------------------------------------------------------
// Stage 2: implicit-GEMM conv. One block per (b, d, h) output row.
// 96 threads = 3 waves; wave ms owns W-positions [ms*16, ms*16+16) and all
// four cout subtiles. Input row (64ch x 52w, zero-padded in w) is staged in
// LDS as bf16 hi/lo once per (kd,kh) and reused by the 5 kw shifts.
// ---------------------------------------------------------------------------
#define AROW 70   // LDS row stride in bf16 elems (35 dwords; gcd(35,64)=1 -> no bank conflicts)

__global__ __launch_bounds__(96) void conv_wmma(
    const float* __restrict__ x,             // [2,64,48,48,48]
    const unsigned int* __restrict__ Bhi,
    const unsigned int* __restrict__ Blo,
    float* __restrict__ out)                 // [2,64,48,48,48]
{
    __shared__ unsigned short Ahi[52 * AROW];
    __shared__ unsigned short Alo[52 * AROW];

    const int h   = blockIdx.x;
    const int d   = blockIdx.y;
    const int b   = blockIdx.z;
    const int tid = threadIdx.x;
    const int lane = tid & 31;
    const int ms   = tid >> 5;               // 0..2 : which 16-wide W subtile

    v8f acc[4] = {v8f{}, v8f{}, v8f{}, v8f{}};

    const int mcol  = lane & 15;             // A: lane -> M row (spatial)
    const int khalf = (lane >> 4) << 3;      // A/B: lanes 16-31 take K+8

    for (int kd = 0; kd < 5; ++kd) {
        const int zz = d + kd - 2;
        if (zz < 0 || zz >= GRID) continue;          // uniform across block
        for (int kh = 0; kh < 5; ++kh) {
            const int yy = h + kh - 2;
            if (yy < 0 || yy >= GRID) continue;      // uniform across block

            __syncthreads();  // previous row fully consumed
            // Stage input row: 64 channels x 52 w-positions, split bf16 hi/lo
            const size_t base0 = (((size_t)(b * CH) * GRID + zz) * GRID + yy) * GRID;
            for (int idx = tid; idx < CH * 52; idx += 96) {
                int ci = idx / 52;
                int w  = idx - ci * 52;
                int wp = w - 2;
                float v = 0.f;
                if (wp >= 0 && wp < GRID)
                    v = x[base0 + (size_t)ci * CH_STRIDE + wp];
                unsigned short hi = f32_to_bf16_rne(v);
                Ahi[w * AROW + ci] = hi;
                Alo[w * AROW + ci] = f32_to_bf16_rne(v - bf16_bits_to_f32(hi));
            }
            __syncthreads();

            for (int kw = 0; kw < 5; ++kw) {
                const int t = (kd * 5 + kh) * 5 + kw;
                const int w = ms * 16 + mcol + kw;   // 0..51, in-staged range
                #pragma unroll
                for (int c = 0; c < 2; ++c) {
                    // A fragments: per-lane gather from LDS (dword-packed K pairs)
                    U32x8 ah, al;
                    #pragma unroll
                    for (int j = 0; j < 8; ++j) {
                        int ci0 = c * 32 + ((j & 4) << 2) + khalf + ((j & 3) << 1);
                        ah.u[j] = *(const unsigned int*)&Ahi[w * AROW + ci0];
                        al.u[j] = *(const unsigned int*)&Alo[w * AROW + ci0];
                    }
                    const unsigned fbase = ((unsigned)t * 8 + (unsigned)c * 4) * FRAG_DW
                                           + ((unsigned)lane << 3);
                    #pragma unroll
                    for (int nt = 0; nt < 4; ++nt) {
                        // B fragments: lane-contiguous, two b128 loads each
                        const uint4* ph = (const uint4*)(Bhi + fbase + nt * FRAG_DW);
                        const uint4* pl = (const uint4*)(Blo + fbase + nt * FRAG_DW);
                        U32x8 bh, bl;
                        bh.q[0] = ph[0]; bh.q[1] = ph[1];
                        bl.q[0] = pl[0]; bl.q[1] = pl[1];
                        // bf16x3: hi*hi + hi*lo + lo*hi, f32 accumulate
                        acc[nt] = __builtin_amdgcn_wmma_f32_16x16x32_bf16(
                            false, ah.v, false, bh.v, (short)0, acc[nt], false, false);
                        acc[nt] = __builtin_amdgcn_wmma_f32_16x16x32_bf16(
                            false, ah.v, false, bl.v, (short)0, acc[nt], false, false);
                        acc[nt] = __builtin_amdgcn_wmma_f32_16x16x32_bf16(
                            false, al.v, false, bh.v, (short)0, acc[nt], false, false);
                    }
                }
            }
        }
    }

    // Epilogue: C/D layout (ISA 7.12.2): lane -> N column, VGPR r -> M row
    // (lanes 16-31 hold M = r + 8).
    const int ncol = lane & 15;
    const int mofs = (lane >> 4) << 3;
    #pragma unroll
    for (int nt = 0; nt < 4; ++nt) {
        const int co = nt * 16 + ncol;
        const size_t obase = (((size_t)(b * CH + co) * GRID + d) * GRID + h) * GRID;
        const int m0 = ms * 16 + mofs;
        #pragma unroll
        for (int r = 0; r < 8; ++r) {
            out[obase + m0 + r] = acc[nt][r];
        }
    }
}

// ---------------------------------------------------------------------------
// Launch
// ---------------------------------------------------------------------------
extern "C" void kernel_launch(void* const* d_in, const int* in_sizes, int n_in,
                              void* d_out, int out_size, void* d_ws, size_t ws_size,
                              hipStream_t stream) {
    const float* x      = (const float*)d_in[0];
    const float* w_sc0  = (const float*)d_in[1];
    const float* w_sc1  = (const float*)d_in[2];
    const float* tp     = (const float*)d_in[3];
    float* out          = (float*)d_out;

    unsigned int* Bhi = (unsigned int*)d_ws;
    unsigned int* Blo = Bhi + B_DWORDS;   // 2 MB total workspace use

    build_kernel_frags<<<NTAP, 256, 0, stream>>>(w_sc0, w_sc1, tp, Bhi, Blo);

    dim3 grid(GRID, GRID, 2);             // (h, d, b)
    conv_wmma<<<grid, 96, 0, stream>>>(x, Bhi, Blo, out);
}
